// GLnL_Module_36009005809724
// MI455X (gfx1250) — matrix operand order repaired and checked by reading, not compiled, not run on hardware
//
#include <hip/hip_runtime.h>

typedef __attribute__((ext_vector_type(16))) _Float16 v16h;
typedef __attribute__((ext_vector_type(8)))  float    v8f;

union Frag { v16h v; uint4 u[2]; };

__device__ __forceinline__ v16h load_frag(const _Float16* p) {
    Frag f;
    f.u[0] = *(const uint4*)(p);
    f.u[1] = *(const uint4*)(p + 8);
    return f.v;
}
// A fragment: 8 halves at base, 8 halves at base+16 (K runs 0..7 / 16..23, hi lanes +8)
__device__ __forceinline__ v16h load_fragA(const _Float16* p) {
    Frag f;
    f.u[0] = *(const uint4*)(p);
    f.u[1] = *(const uint4*)(p + 16);
    return f.v;
}

__device__ __forceinline__ v8f wmma16(v16h a, v16h b, v8f c) {
    return __builtin_amdgcn_wmma_f32_16x16x32_f16(false, a, false, b, (short)0, c, false, false);
}

#define ZERO8 {0.f,0.f,0.f,0.f,0.f,0.f,0.f,0.f}

// ---------------- sizes ----------------
// B=4, C=128, I=64, H=W=64, N=4096
#define NB 4
#define CC 128
#define II 64
#define NN 4096
// workspace half-offsets
#define WSZ1_OFF 0                 // 6*4*4*512 = 49152
#define WSZWL_OFF 49152            // 2*4*512   = 4096
#define WSZW_OFF 53248             // 2*8*512   = 8192
#define PROJ_OFF 61440             // 6*4*4096*64
#define ATTN_OFF (61440 + 6291456) // 2*4*4096*64

// ------------------------------------------------------------------
// Weight swizzle into WMMA B-operand per-lane layout:
// B (K x 16): lane = column (lane&15); lanes<16 hold K 0..7,16..23,
// lanes>=16 hold K 8..15,24..31 (per 32-K chunk).
// ------------------------------------------------------------------
__global__ __launch_bounds__(256) void k_prep(
    const float* __restrict__ wt_nl, const float* __restrict__ wp_nl,
    const float* __restrict__ wt_l,  const float* __restrict__ wp_l,
    const float* __restrict__ wg_nl, const float* __restrict__ wg_l,
    const float* __restrict__ wW,    const float* __restrict__ wWl,
    _Float16* __restrict__ ws)
{
    int idx = blockIdx.x * 256 + threadIdx.x;
    int t    = idx & 15;
    int lane = (idx >> 4) & 31;
    int blk  = idx >> 9;
    int col  = lane & 15;
    int kl   = ((lane < 16) ? 0 : 8) + ((t < 8) ? t : t + 8);
    if (idx < 49152) {
        // stage-1 weights: 6 mats [I=64, C=128]; blocks: ((mat*4+kb)*4+ib)
        int ib = blk & 3, kb = (blk >> 2) & 3, mat = blk >> 4;
        const float* w = (mat == 0) ? wt_nl : (mat == 1) ? wp_nl :
                         (mat == 2) ? wt_l  : (mat == 3) ? wp_l  :
                         (mat == 4) ? wg_nl : wg_l;
        ws[WSZ1_OFF + idx] = (_Float16)w[(ib * 16 + col) * CC + kb * 32 + kl];
    } else if (idx < 53248) {
        int j = idx - 49152;
        int b2 = j >> 9;                 // kb*4 + ib
        int ib = b2 & 3, kb = b2 >> 2;   // kb in 0..1
        ws[idx] = (_Float16)wWl[(ib * 16 + col) * II + kb * 32 + kl];
    } else if (idx < 61440) {
        int j = idx - 53248;
        int b2 = j >> 9;                 // kb*8 + cb
        int cb = b2 & 7, kb = b2 >> 3;   // kb in 0..1
        ws[idx] = (_Float16)wW[(cb * 16 + col) * II + kb * 32 + kl];
    }
}

// ------------------------------------------------------------------
// Stage 1: projections.
//   mats 0..3 (theta/phi): out [mat][b][n][i] f16   (= attention Q/K rows)
//   mats 4..5 (g):         out [mat][b][i][n] f16   (= V^T, fed straight to P.V)
// ------------------------------------------------------------------
#define XSTR 136
__global__ __launch_bounds__(256) void k_proj(
    const float* __restrict__ x, const float* __restrict__ mask,
    const float* __restrict__ bt_nl, const float* __restrict__ bp_nl,
    const float* __restrict__ bt_l,  const float* __restrict__ bp_l,
    const float* __restrict__ bg_nl, const float* __restrict__ bg_l,
    const _Float16* __restrict__ wsz, _Float16* __restrict__ proj)
{
    __shared__ _Float16 xl[128 * XSTR];
    __shared__ float ml[128];
    const int tid = threadIdx.x;
    const int b  = blockIdx.x >> 5;
    const int n0 = (blockIdx.x & 31) * 128;
    const float* xb = x + (size_t)b * CC * NN;

    for (int u = tid; u < 4096; u += 256) {          // 4096 float4 = 128x128 f32
        int c = u >> 5, j4 = (u & 31) * 4;
        float4 v = *(const float4*)(xb + (size_t)c * NN + n0 + j4);
        xl[(j4 + 0) * XSTR + c] = (_Float16)v.x;
        xl[(j4 + 1) * XSTR + c] = (_Float16)v.y;
        xl[(j4 + 2) * XSTR + c] = (_Float16)v.z;
        xl[(j4 + 3) * XSTR + c] = (_Float16)v.w;
    }
    if (tid < 128) ml[tid] = mask[(size_t)b * NN + n0 + tid];
    __syncthreads();

    const int lane = tid & 31, w = tid >> 5;
    const int lo = lane & 15, hi8 = (lane >> 4) * 8;
    const int p0 = w * 16;

    v16h a[4];
#pragma unroll
    for (int kc = 0; kc < 4; ++kc)
        a[kc] = load_fragA(&xl[(p0 + lo) * XSTR + kc * 32 + hi8]);

    const float* biases[6] = {bt_nl, bp_nl, bt_l, bp_l, bg_nl, bg_l};
#pragma unroll
    for (int mat = 0; mat < 6; ++mat) {
        const float* bias = biases[mat];
        const bool domask = (mat == 2) || (mat == 3);
#pragma unroll
        for (int ib = 0; ib < 4; ++ib) {
            v8f acc = ZERO8;
#pragma unroll
            for (int kc = 0; kc < 4; ++kc) {
                v16h bf = load_frag(wsz + ((((mat * 4 + kc) * 4 + ib) << 9) + lane * 16));
                acc = wmma16(a[kc], bf, acc);
            }
            float bv = bias[ib * 16 + lo];
            if (mat < 4) {
                _Float16* pout = proj + ((size_t)(mat * 4 + b) * NN + (n0 + p0 + hi8)) * II
                                      + ib * 16 + lo;
#pragma unroll
                for (int r = 0; r < 8; ++r) {
                    float v = acc[r] + bv;
                    if (domask) v *= ml[p0 + r + hi8];
                    pout[(size_t)r * II] = (_Float16)v;
                }
            } else {
                // g stored transposed [i][n]: 8 consecutive n per lane -> one b128 store
                union { _Float16 h[8]; uint4 u; } pk;
#pragma unroll
                for (int r = 0; r < 8; ++r) pk.h[r] = (_Float16)(acc[r] + bv);
                *(uint4*)(proj + (size_t)(mat * 4 + b) * NN * II
                               + (size_t)(ib * 16 + lo) * NN + (n0 + p0 + hi8)) = pk.u;
            }
        }
    }
}

// ------------------------------------------------------------------
// Stage 2: flash attention. Q/K = [N,64] f16, V^T = [64,N] f16.
// K tiles double-buffered in LDS via async global->LDS loads (ASYNCcnt),
// V fragments read straight from L2 in B-operand order.
// ------------------------------------------------------------------
#define KSTR 72
__global__ __launch_bounds__(256) void k_attn(
    const _Float16* __restrict__ proj, _Float16* __restrict__ aout)
{
    __shared__ _Float16 kl[2][64 * KSTR];
    __shared__ _Float16 pl[8 * 16 * KSTR];
    const int tid = threadIdx.x;
    const int br = blockIdx.x >> 7;
    const int b  = (blockIdx.x >> 5) & 3;
    const int q0 = (blockIdx.x & 31) * 128;
    const int qm = br ? 2 : 0, km = br ? 3 : 1, vm = br ? 5 : 4;
    const _Float16* Q  = proj + (size_t)(qm * 4 + b) * NN * II;
    const _Float16* K  = proj + (size_t)(km * 4 + b) * NN * II;
    const _Float16* Vt = proj + (size_t)(vm * 4 + b) * NN * II;   // [i][n]
    const int lane = tid & 31, w = tid >> 5;
    const int lo = lane & 15, hi8 = (lane >> 4) * 8, hi16 = (lane >> 4) * 16;
    const int pw = w * 16 * KSTR;

    const unsigned klbase = (unsigned)(size_t)&kl[0][0];
    const unsigned long long kbase = (unsigned long long)(size_t)K;

    // async-stage one 64-key K tile (8 KB) into buffer (kb2 & 1)
    auto issue = [&](int kb2) {
        const unsigned lbase = klbase + (unsigned)((kb2 & 1) * 64 * KSTR * 2);
        const int m0i = kb2 * 64;
#pragma unroll
        for (int uu = 0; uu < 2; ++uu) {
            int u = tid + uu * 256;          // 512 x 16B = 8 KB
            int m = u >> 3, q = u & 7;
            unsigned laddr = lbase + (unsigned)(m * (KSTR * 2) + q * 16);
            unsigned voff  = (unsigned)(((m0i + m) * II + q * 8) * 2);
            asm volatile("global_load_async_to_lds_b128 %0, %1, %2 offset:0"
                         :: "v"(laddr), "v"(voff), "s"(kbase) : "memory");
        }
    };

    v16h qa[2];
    {
        const _Float16* qp = Q + (size_t)(q0 + w * 16 + lo) * II + hi8;
        qa[0] = load_fragA(qp);
        qa[1] = load_fragA(qp + 32);
    }
    v8f o[4] = {ZERO8, ZERO8, ZERO8, ZERO8};
    float mrow[8], lrow[8];
#pragma unroll
    for (int r = 0; r < 8; ++r) { mrow[r] = -1e30f; lrow[r] = 0.f; }

    issue(0);
    for (int kb = 0; kb < NN / 64; ++kb) {
        const int m0 = kb * 64;
        asm volatile("s_wait_asynccnt 0x0" ::: "memory");  // my tile writes done
        __syncthreads();                                   // everyone's done, prev readers done
        if (kb + 1 < NN / 64) issue(kb + 1);               // prefetch next tile under compute
        const _Float16* kbuf = &kl[kb & 1][0];

        v8f s[4];
#pragma unroll
        for (int mt = 0; mt < 4; ++mt) {
            v8f acc = ZERO8;
#pragma unroll
            for (int kc = 0; kc < 2; ++kc) {
                v16h bf = load_fragA(&kbuf[(mt * 16 + lo) * KSTR + kc * 32 + hi8]);
                acc = wmma16(qa[kc], bf, acc);
            }
            s[mt] = acc;
        }
        // online softmax over this 64-key block
#pragma unroll
        for (int r = 0; r < 8; ++r) {
            float t = fmaxf(fmaxf(s[0][r], s[1][r]), fmaxf(s[2][r], s[3][r]));
            t = fmaxf(t, __shfl_xor(t, 1, 32));
            t = fmaxf(t, __shfl_xor(t, 2, 32));
            t = fmaxf(t, __shfl_xor(t, 4, 32));
            t = fmaxf(t, __shfl_xor(t, 8, 32));
            float nm = fmaxf(mrow[r], t);
            float sc = __expf(mrow[r] - nm);
            mrow[r] = nm;
            float ps = 0.f;
#pragma unroll
            for (int mt = 0; mt < 4; ++mt) {
                float p = __expf(s[mt][r] - nm);
                ps += p;
                pl[pw + (r + hi8) * KSTR + mt * 16 + lo] = (_Float16)p;
            }
            ps += __shfl_xor(ps, 1, 32);
            ps += __shfl_xor(ps, 2, 32);
            ps += __shfl_xor(ps, 4, 32);
            ps += __shfl_xor(ps, 8, 32);
            lrow[r] = lrow[r] * sc + ps;
#pragma unroll
            for (int it = 0; it < 4; ++it) o[it][r] *= sc;
        }
        asm volatile("" ::: "memory");   // keep P stores before P reloads
        v16h pa[2];
        pa[0] = load_fragA(&pl[pw + lo * KSTR + hi8]);
        pa[1] = load_fragA(&pl[pw + lo * KSTR + 32 + hi8]);
#pragma unroll
        for (int it = 0; it < 4; ++it) {
#pragma unroll
            for (int kc = 0; kc < 2; ++kc) {
                v16h bf = load_fragA(Vt + (size_t)(it * 16 + lo) * NN + m0 + kc * 32 + hi8);
                o[it] = wmma16(pa[kc], bf, o[it]);
            }
        }
    }
    _Float16* ob = aout + ((size_t)(br * 4 + b) * NN + q0 + w * 16 + hi8) * II + lo;
#pragma unroll
    for (int r = 0; r < 8; ++r) {
        float inv = 1.f / lrow[r];
#pragma unroll
        for (int it = 0; it < 4; ++it)
            ob[(size_t)r * II + it * 16] = (_Float16)(o[it][r] * inv);
    }
}

// ------------------------------------------------------------------
// Stage 3: y_l -> wWl conv + ReLU + BN; add y_nl; wW conv + bias + ReLU + BN + x
// ------------------------------------------------------------------
__global__ __launch_bounds__(256) void k_final(
    const _Float16* __restrict__ aout,
    const _Float16* __restrict__ wszWl, const _Float16* __restrict__ wszW,
    const float* __restrict__ gWl, const float* __restrict__ betaWl,
    const float* __restrict__ bW,  const float* __restrict__ gW,
    const float* __restrict__ betaW, const float* __restrict__ x,
    float* __restrict__ out)
{
    __shared__ _Float16 yl[8 * 16 * KSTR];
    const int tid = threadIdx.x;
    const int b  = blockIdx.x >> 5;
    const int n0 = (blockIdx.x & 31) * 128;
    const int lane = tid & 31, w = tid >> 5;
    const int lo = lane & 15, hi8 = (lane >> 4) * 8;
    const int pw = w * 16 * KSTR;
    const int nbase = n0 + w * 16;
    const _Float16* An = aout + (size_t)(0 + b) * NN * II;
    const _Float16* Al = aout + (size_t)(4 + b) * NN * II;
    const float rs = rsqrtf(1.f + 1e-5f);

    v16h al[2];
    {
        const _Float16* p = Al + (size_t)(nbase + lo) * II + hi8;
        al[0] = load_fragA(p);
        al[1] = load_fragA(p + 32);
    }
#pragma unroll
    for (int ib = 0; ib < 4; ++ib) {
        v8f acc = ZERO8;
#pragma unroll
        for (int kc = 0; kc < 2; ++kc) {
            v16h bf = load_frag(wszWl + (((kc * 4 + ib) << 9) + lane * 16));
            acc = wmma16(al[kc], bf, acc);
        }
        int ci = ib * 16 + lo;
        float scl = gWl[ci] * rs;
        float bet = betaWl[ci];
        const _Float16* anp = An + (size_t)(nbase + hi8) * II + ci;
#pragma unroll
        for (int r = 0; r < 8; ++r) {
            float v = fmaxf(acc[r], 0.f) * scl + bet + (float)anp[(size_t)r * II];
            yl[pw + (r + hi8) * KSTR + ci] = (_Float16)v;
        }
    }
    asm volatile("" ::: "memory");
    v16h ya[2];
    ya[0] = load_fragA(&yl[pw + lo * KSTR + hi8]);
    ya[1] = load_fragA(&yl[pw + lo * KSTR + 32 + hi8]);
#pragma unroll
    for (int cb = 0; cb < 8; ++cb) {
        v8f acc = ZERO8;
#pragma unroll
        for (int kc = 0; kc < 2; ++kc) {
            v16h bf = load_frag(wszW + (((kc * 8 + cb) << 9) + lane * 16));
            acc = wmma16(ya[kc], bf, acc);
        }
        int c = cb * 16 + lo;
        float bw = bW[c], scl = gW[c] * rs, bet = betaW[c];
        const float* xp = x + ((size_t)b * CC + c) * NN + nbase + hi8;
        float* op = out + ((size_t)b * CC + c) * NN + nbase + hi8;
#pragma unroll
        for (int r = 0; r < 8; ++r)
            op[r] = fmaxf(acc[r] + bw, 0.f) * scl + bet + xp[r];
    }
}

extern "C" void kernel_launch(void* const* d_in, const int* in_sizes, int n_in,
                              void* d_out, int out_size, void* d_ws, size_t ws_size,
                              hipStream_t stream) {
    const float* x      = (const float*)d_in[0];
    const float* mask   = (const float*)d_in[1];
    const float* wt_nl  = (const float*)d_in[2];
    const float* bt_nl  = (const float*)d_in[3];
    const float* wp_nl  = (const float*)d_in[4];
    const float* bp_nl  = (const float*)d_in[5];
    const float* wt_l   = (const float*)d_in[6];
    const float* bt_l   = (const float*)d_in[7];
    const float* wp_l   = (const float*)d_in[8];
    const float* bp_l   = (const float*)d_in[9];
    const float* wg_nl  = (const float*)d_in[10];
    const float* bg_nl  = (const float*)d_in[11];
    const float* wg_l   = (const float*)d_in[12];
    const float* bg_l   = (const float*)d_in[13];
    const float* wW     = (const float*)d_in[14];
    const float* bW     = (const float*)d_in[15];
    const float* gW     = (const float*)d_in[16];
    const float* betaW  = (const float*)d_in[17];
    const float* wWl    = (const float*)d_in[18];
    const float* gWl    = (const float*)d_in[19];
    const float* betaWl = (const float*)d_in[20];

    _Float16* ws    = (_Float16*)d_ws;
    _Float16* wsz1  = ws + WSZ1_OFF;
    _Float16* wszWl = ws + WSZWL_OFF;
    _Float16* wszW  = ws + WSZW_OFF;
    _Float16* proj  = ws + PROJ_OFF;
    _Float16* attn  = ws + ATTN_OFF;

    k_prep<<<240, 256, 0, stream>>>(wt_nl, wp_nl, wt_l, wp_l, wg_nl, wg_l, wW, wWl, ws);
    k_proj<<<NB * (NN / 128), 256, 0, stream>>>(x, mask, bt_nl, bp_nl, bt_l, bp_l,
                                                bg_nl, bg_l, wsz1, proj);
    k_attn<<<2 * NB * (NN / 128), 256, 0, stream>>>(proj, attn);
    k_final<<<NB * (NN / 128), 256, 0, stream>>>(attn, wszWl, wszW, gWl, betaWl,
                                                 bW, gW, betaW, x, (float*)d_out);
}
